// NUSpectralConv2d_57140244906257
// MI455X (gfx1250) — matrix operand order repaired and checked by reading, not compiled, not run on hardware
//
#include <hip/hip_runtime.h>
#include <math.h>

// ---- problem constants ----
#define BB     4
#define CINC   32
#define COUTC  32
#define SSEG   8
#define NPTS   16384
#define MPSEG  2048
#define HDIM   64
#define WDIM   64
#define M1C    12
#define M2C    12
#define HIMGC  25
#define WIMGC  25
#define SCOUT  256
#define NFY    13     // ky index 0..12  <->  ky = -12..0
#define NFX    25     // kx index 0..24  <->  kx = -12..12
#define KHW    625    // 25*25 angle (hw) pairs
#define KTOT   1280   // 2*625 padded to 40 k-steps of 32
#define KSTEPS 40
#define EPSF   1e-6f
#define PI_F   3.14159265358979f
#define TWO_PI_F 6.28318530717959f

typedef __attribute__((ext_vector_type(16))) _Float16 v16h;
typedef __attribute__((ext_vector_type(8)))  float    v8f;

// ---------------------------------------------------------------------------
// Kernel 0: per-(b,s) min/max over the gathered segment points -> om angles
// ---------------------------------------------------------------------------
__global__ __launch_bounds__(256) void k_om(const float* __restrict__ loc,
                                            const int* __restrict__ ind,
                                            float* __restrict__ om) {
  const int bs = blockIdx.x;           // 0..31
  const int b = bs >> 3, s = bs & 7;
  const int tid = threadIdx.x;
  __shared__ float red[4][256];
  float mnx = 1e30f, mxx = -1e30f, mny = 1e30f, mxy = -1e30f;
  const int base = b * NPTS + s * MPSEG;
  for (int m = tid; m < MPSEG; m += 256) {
    int n = ind[base + m];
    float px = loc[(b * NPTS + n) * 2 + 0];
    float py = loc[(b * NPTS + n) * 2 + 1];
    mnx = fminf(mnx, px); mxx = fmaxf(mxx, px);
    mny = fminf(mny, py); mxy = fmaxf(mxy, py);
  }
  red[0][tid] = mnx; red[1][tid] = mxx; red[2][tid] = mny; red[3][tid] = mxy;
  __syncthreads();
  for (int off = 128; off > 0; off >>= 1) {
    if (tid < off) {
      red[0][tid] = fminf(red[0][tid], red[0][tid + off]);
      red[1][tid] = fmaxf(red[1][tid], red[1][tid + off]);
      red[2][tid] = fminf(red[2][tid], red[2][tid + off]);
      red[3][tid] = fmaxf(red[3][tid], red[3][tid + off]);
    }
    __syncthreads();
  }
  const float gmnx = red[0][0], gmxx = red[1][0];
  const float gmny = red[2][0], gmxy = red[3][0];
  const float sx = TWO_PI_F / (gmxx - gmnx + EPSF);
  const float sy = TWO_PI_F / (gmxy - gmny + EPSF);
  for (int m = tid; m < MPSEG; m += 256) {
    int n = ind[base + m];
    float px = loc[(b * NPTS + n) * 2 + 0];
    float py = loc[(b * NPTS + n) * 2 + 1];
    om[(bs * MPSEG + m) * 2 + 0] = (px - gmnx) * sx - PI_F;
    om[(bs * MPSEG + m) * 2 + 1] = (py - gmny) * sy - PI_F;
  }
}

// ---------------------------------------------------------------------------
// Kernel 1a: partial DFT along W. G[b,i,h,ky] for ky = -12..0 (13 bins)
// ---------------------------------------------------------------------------
__global__ __launch_bounds__(256) void k_dft_w(const float* __restrict__ x,
                                               float* __restrict__ Gre,
                                               float* __restrict__ Gim) {
  const int TOT = BB * CINC * HDIM * NFY;
  int idx = blockIdx.x * 256 + threadIdx.x;
  if (idx >= TOT) return;
  int ky = idx % NFY; int r = idx / NFY;
  int h = r % HDIM; r /= HDIM;
  int i = r % CINC; int b = r / CINC;
  const float w0 = -TWO_PI_F * (float)(ky - 12) / 65.0f;  // e^{-2πi k w /65}
  const float* xp = x + (((long)(b * CINC + i)) * HDIM + h) * WDIM;
  float ar = 0.f, ai = 0.f;
  for (int w = 0; w < WDIM; ++w) {
    float sn, cs;
    __sincosf(w0 * (float)w, &sn, &cs);
    float xv = xp[w];
    ar += xv * cs;
    ai += xv * sn;
  }
  const int o = ((b * CINC + i) * HDIM + h) * NFY + ky;
  Gre[o] = ar; Gim[o] = ai;
}

// ---------------------------------------------------------------------------
// Kernel 1b: partial DFT along H. F[b,i,kx,ky], kx=-12..12, ortho norm 1/65
// ---------------------------------------------------------------------------
__global__ __launch_bounds__(256) void k_dft_h(const float* __restrict__ Gre,
                                               const float* __restrict__ Gim,
                                               float* __restrict__ Fre,
                                               float* __restrict__ Fim) {
  const int TOT = BB * CINC * NFX * NFY;
  int idx = blockIdx.x * 256 + threadIdx.x;
  if (idx >= TOT) return;
  int ky = idx % NFY; int r = idx / NFY;
  int kx = r % NFX; r /= NFX;
  int i = r % CINC; int b = r / CINC;
  const float w0 = -TWO_PI_F * (float)(kx - 12) / 65.0f;
  float ar = 0.f, ai = 0.f;
  const int gbase = (b * CINC + i) * HDIM;
  for (int h = 0; h < HDIM; ++h) {
    float sn, cs;
    __sincosf(w0 * (float)h, &sn, &cs);
    float gr = Gre[(gbase + h) * NFY + ky];
    float gi = Gim[(gbase + h) * NFY + ky];
    ar += gr * cs - gi * sn;
    ai += gr * sn + gi * cs;
  }
  const int o = ((b * CINC + i) * NFX + kx) * NFY + ky;
  const float nrm = 1.0f / 65.0f;   // ortho: 1/sqrt(65*65)
  Fre[o] = ar * nrm; Fim[o] = ai * nrm;
}

// ---------------------------------------------------------------------------
// zero-fill (covers the K padding lanes of Afrag: k in [1250,1280))
// ---------------------------------------------------------------------------
__global__ __launch_bounds__(256) void k_zero(unsigned int* __restrict__ p, int nwords) {
  int idx = blockIdx.x * 256 + threadIdx.x;
  if (idx < nwords) p[idx] = 0u;
}

// ---------------------------------------------------------------------------
// Kernel 2: channel mix + Hermitian assembly, written directly in the
// per-lane-contiguous WMMA A-fragment layout (ISA §7.12.2, 16-bit A 16x32):
//   k_local = (t<8 ? t : t+8) + (laneHi ? 8 : 0)
// Interleaved K mapping: k = 2*hw + part; part 0 -> Re(imgs) (pairs with cos),
// part 1 -> -Im(imgs) (pairs with sin). hw >= 625 padded to zero.
// ---------------------------------------------------------------------------
__global__ __launch_bounds__(256) void k_build_A(const float* __restrict__ Fre,
                                                 const float* __restrict__ Fim,
                                                 const float* __restrict__ w1_re,
                                                 const float* __restrict__ w1_im,
                                                 const float* __restrict__ w2_re,
                                                 const float* __restrict__ w2_im,
                                                 _Float16* __restrict__ Afrag) {
  const int TOT = BB * SCOUT * KHW;   // 640000
  int idx = blockIdx.x * 256 + threadIdx.x;
  if (idx >= TOT) return;
  int hw = idx % KHW; int r = idx / KHW;
  int o = r % SCOUT; int b = r / SCOUT;
  int h = hw / WIMGC, w = hw % WIMGC;

  float vre = 0.f, vim = 0.f;
  if (w != M2C) {
    // im1 region (direct for w<12, Hermitian-conjugate mirror for w>12)
    int hs = (w < M2C) ? h : (24 - h);
    int ws = (w < M2C) ? w : (24 - w);       // 0..11 == ky index
    float ar = 0.f, ai = 0.f;
    for (int i = 0; i < CINC; ++i) {
      int fo = ((b * CINC + i) * NFX + hs) * NFY + ws;
      float fr = Fre[fo], fi = Fim[fo];
      int wo = ((i * SCOUT + o) * NFX + hs) * M2C + ws;
      float wr = w1_re[wo], wi = w1_im[wo];
      ar += fr * wr - fi * wi;
      ai += fr * wi + fi * wr;
    }
    vre = ar;
    vim = (w < M2C) ? ai : -ai;
  } else {
    // center column: im2 (rows 0..12), conj mirror (rows 13..24), real at (12,12)
    int xs = (h <= M1C) ? h : (24 - h);      // 0..12 == kx index, ky=0 -> NFY-1
    float ar = 0.f, ai = 0.f;
    for (int i = 0; i < CINC; ++i) {
      int fo = ((b * CINC + i) * NFX + xs) * NFY + 12;
      float fr = Fre[fo], fi = Fim[fo];
      int wo = (i * SCOUT + o) * (M1C + 1) + xs;
      float wr = w2_re[wo], wi = w2_im[wo];
      ar += fr * wr - fi * wi;
      ai += fr * wi + fi * wr;
    }
    vre = ar;
    vim = (h <= M1C) ? ai : -ai;
    if (h == M1C) vim = 0.f;
  }

  const int s = o >> 5, c = o & 31;
  const int bs = b * SSEG + s;
  const int mtile = c >> 4;
#pragma unroll
  for (int part = 0; part < 2; ++part) {
    int k = 2 * hw + part;                 // interleaved: even=Re/cos, odd=-Im/sin
    float val = (part == 0) ? vre : -vim;
    int kstep = k >> 5;
    int klocal = k & 31;
    int hi = (klocal >> 3) & 1;
    int t = (klocal & 7) + ((klocal >= 16) ? 8 : 0);
    int lane = hi * 16 + (c & 15);
    long off = ((((long)(bs * KSTEPS + kstep)) * 2 + mtile) * 32 + lane) * 16 + t;
    Afrag[off] = (_Float16)val;
  }
}

// ---------------------------------------------------------------------------
// Kernel 3: the NUDFT evaluation as 32 real GEMMs (32 x 1280 x 2048) on the
// f16 WMMA path. Each wave owns a 32(c) x 16(m) tile (2 accumulators).
// B fragments are generated in-register, branchless: with the interleaved K
// mapping each angle feeds two adjacent slots (cos, sin) from one __sincosf.
// Epilogue: /25 (sqrt(625)), *weights + bias, scatter through ind.
// ---------------------------------------------------------------------------
__global__ __launch_bounds__(256) void k_nudft_wmma(const _Float16* __restrict__ Afrag,
                                                    const float* __restrict__ om,
                                                    const int* __restrict__ ind,
                                                    const float* __restrict__ weights,
                                                    const float* __restrict__ bias,
                                                    float* __restrict__ out) {
  const int bs = blockIdx.y;           // 0..31 -> (b,s)
  const int b = bs >> 3, s = bs & 7;
  const int wave = threadIdx.x >> 5;   // 8 waves per block
  const int lane = threadIdx.x & 31;
  const int nloc = lane & 15;
  const int laneHi = lane >> 4;
  const int mpt = blockIdx.x * 128 + wave * 16 + nloc;   // point within segment

  const float om0 = om[(bs * MPSEG + mpt) * 2 + 0];
  const float om1 = om[(bs * MPSEG + mpt) * 2 + 1];
  const float angBase = -12.f * (om0 + om1);   // ang = om0*hm + om1*wm + angBase

  const v16h* __restrict__ abase =
      (const v16h*)Afrag + (long)bs * (KSTEPS * 2 * 32);

  v8f acc0 = {};
  v8f acc1 = {};
  // B layout: lanes 0-15 hold K 0-7,16-23, lanes 16-31 hold K 8-15,24-31 of each k-step.
  const int pairBase = laneHi * 4;     // angle-pair index offset within k-step

  for (int kstep = 0; kstep < KSTEPS; ++kstep) {
    v16h a0 = abase[(kstep * 2 + 0) * 32 + lane];
    v16h a1 = abase[(kstep * 2 + 1) * 32 + lane];
    if (kstep + 1 < KSTEPS)
      __builtin_prefetch((const void*)&abase[((kstep + 1) * 2) * 32 + lane], 0, 3);

    v16h bfrag;
#pragma unroll
    for (int tp = 0; tp < 8; ++tp) {
      int hw = kstep * 16 + pairBase + tp + ((tp >= 4) ? 4 : 0);     // angle (hw) index, < 640
      int hm = hw / 25;
      int wm = hw - hm * 25;
      float ang = fmaf(om0, (float)hm, fmaf(om1, (float)wm, angBase));
      float sn, cs;
      __sincosf(ang, &sn, &cs);
      float ok = (hw < KHW) ? 1.f : 0.f;       // branchless pad mask
      bfrag[2 * tp]     = (_Float16)(cs * ok);
      bfrag[2 * tp + 1] = (_Float16)(sn * ok);
    }

    acc0 = __builtin_amdgcn_wmma_f32_16x16x32_f16(false, a0, false, bfrag,
                                                  (short)0, acc0, false, false);
    acc1 = __builtin_amdgcn_wmma_f32_16x16x32_f16(false, a1, false, bfrag,
                                                  (short)0, acc1, false, false);
  }

  const int nIdx = ind[b * NPTS + s * MPSEG + mpt];
  const float inv25 = 1.0f / 25.0f;    // 1/sqrt(HIMG*WIMG)
#pragma unroll
  for (int r = 0; r < 8; ++r) {
    // C/D layout: VGPR r holds row M = r (lanes 0-15) or M = 8+r (lanes 16-31)
    int c0 = r + laneHi * 8;           // rows 0..15  (tile 0)
    int c1 = 16 + c0;                  // rows 16..31 (tile 1)
    float v0 = acc0[r] * inv25 * weights[s * COUTC + c0] + bias[s * COUTC + c0];
    float v1 = acc1[r] * inv25 * weights[s * COUTC + c1] + bias[s * COUTC + c1];
    out[((long)(b * COUTC + c0)) * NPTS + nIdx] = v0;
    out[((long)(b * COUTC + c1)) * NPTS + nIdx] = v1;
  }
}

// ---------------------------------------------------------------------------
extern "C" void kernel_launch(void* const* d_in, const int* in_sizes, int n_in,
                              void* d_out, int out_size, void* d_ws, size_t ws_size,
                              hipStream_t stream) {
  const float* x       = (const float*)d_in[0];
  const float* loc     = (const float*)d_in[1];
  const int*   ind     = (const int*)d_in[2];
  // d_in[3] (sep) is unused: segments are uniform N/S, matching the reference.
  const float* weights = (const float*)d_in[4];
  const float* bias    = (const float*)d_in[5];
  const float* w1_re   = (const float*)d_in[6];
  const float* w1_im   = (const float*)d_in[7];
  const float* w2_re   = (const float*)d_in[8];
  const float* w2_im   = (const float*)d_in[9];
  float* out = (float*)d_out;

  char* ws = (char*)d_ws;
  size_t off = 0;
  auto walloc = [&](size_t bytes) -> void* {
    void* p = (void*)(ws + off);
    off += (bytes + 255) & ~(size_t)255;
    return p;
  };
  float*    om    = (float*)walloc((size_t)BB * SSEG * MPSEG * 2 * sizeof(float));
  float*    Gre   = (float*)walloc((size_t)BB * CINC * HDIM * NFY * sizeof(float));
  float*    Gim   = (float*)walloc((size_t)BB * CINC * HDIM * NFY * sizeof(float));
  float*    Fre   = (float*)walloc((size_t)BB * CINC * NFX * NFY * sizeof(float));
  float*    Fim   = (float*)walloc((size_t)BB * CINC * NFX * NFY * sizeof(float));
  _Float16* Afrag = (_Float16*)walloc((size_t)32 * KSTEPS * 2 * 32 * 16 * sizeof(_Float16));
  (void)ws_size; (void)in_sizes; (void)n_in; (void)out_size;

  // 0) segment angles
  k_om<<<dim3(32), dim3(256), 0, stream>>>(loc, ind, om);

  // 1) separable partial DFT (only the 325 needed bins)
  {
    int tot = BB * CINC * HDIM * NFY;
    k_dft_w<<<dim3((tot + 255) / 256), dim3(256), 0, stream>>>(x, Gre, Gim);
  }
  {
    int tot = BB * CINC * NFX * NFY;
    k_dft_h<<<dim3((tot + 255) / 256), dim3(256), 0, stream>>>(Gre, Gim, Fre, Fim);
  }

  // 2) zero-fill (K padding) + build swizzled f16 A fragments
  {
    int nwords = (32 * KSTEPS * 2 * 32 * 16) / 2;
    k_zero<<<dim3((nwords + 255) / 256), dim3(256), 0, stream>>>((unsigned int*)Afrag, nwords);
    int tot = BB * SCOUT * KHW;
    k_build_A<<<dim3((tot + 255) / 256), dim3(256), 0, stream>>>(
        Fre, Fim, w1_re, w1_im, w2_re, w2_im, Afrag);
  }

  // 3) WMMA NUDFT + affine + scatter
  k_nudft_wmma<<<dim3(16, 32), dim3(256), 0, stream>>>(Afrag, om, ind, weights, bias, out);
}
